// Model_49452253446781
// MI455X (gfx1250) — matrix operand order repaired and checked by reading, not compiled, not run on hardware
//
#include <hip/hip_runtime.h>
#include <hip/hip_bf16.h>

// ---------------------------------------------------------------------------
// Problem constants (from the reference)
// ---------------------------------------------------------------------------
constexpr int NB = 32, NL = 1024, NC = 21, ND = 512, NH = 8, NDF = 2048, NE = 3, NP = 336;

typedef __bf16 bf16;
typedef __attribute__((ext_vector_type(16))) __bf16 v16bf;
typedef __attribute__((ext_vector_type(8)))  __bf16 v8bf;
typedef __attribute__((ext_vector_type(8)))  float  v8f;
typedef __attribute__((ext_vector_type(4)))  int    v4i;

union AFrag { v16bf v; v8bf h[2]; };

// float <-> bf16 via integer ops (round-to-nearest-even)
__device__ __forceinline__ bf16 f2b(float x) {
    union { float f; unsigned u; } a; a.f = x;
    unsigned r = a.u + 0x7FFFu + ((a.u >> 16) & 1u);
    union { unsigned short s; bf16 b; } o; o.s = (unsigned short)(r >> 16);
    return o.b;
}

// ---------------------------------------------------------------------------
// Async global->LDS staging (GLOBAL_LOAD_ASYNC_TO_LDS_B128, ASYNCcnt) when the
// toolchain exposes it; synchronous copy fallback otherwise.  Probe round 2
// showed the builtin takes generic `int4*` pointers (no explicit addrspace).
// ---------------------------------------------------------------------------
#if __has_builtin(__builtin_amdgcn_global_load_async_to_lds_b128) && \
    __has_builtin(__builtin_amdgcn_s_wait_asynccnt)
#define USE_ASYNC_LDS 1
#else
#define USE_ASYNC_LDS 0
#endif

#if USE_ASYNC_LDS
#define ASYNC_CP16(dst, src) \
    __builtin_amdgcn_global_load_async_to_lds_b128((v4i*)(void*)(src), (v4i*)(void*)(dst), 0, 0)
#define WAIT_ASYNC(n) __builtin_amdgcn_s_wait_asynccnt(n)
#else
#define ASYNC_CP16(dst, src) \
    (*reinterpret_cast<float4*>(dst) = *reinterpret_cast<const float4*>(src))
#define WAIT_ASYNC(n)
#endif

// ---------------------------------------------------------------------------
// Generic bf16 GEMM:  D[M,N] = epilogue( A[M,K] * B[K,N] )
//   A: row-major (stride sAm, unit k-stride), B: row-major (stride sBk, unit n)
//   Block tile 128x128, BK=32, 8 waves (2x4), each wave 64x32 -> 8 WMMA frags.
//   Double-buffered LDS; A staged via async-LDS, B staged transposed so both
//   fragment reads are contiguous ds_load_b128.
// ---------------------------------------------------------------------------
__global__ __launch_bounds__(256)
void gemm_bf16_kernel(const bf16* __restrict__ A, long sAm, long sAb,
                      const bf16* __restrict__ Bm, long sBk, long sBb,
                      float* __restrict__ Df, bf16* __restrict__ Dh,
                      long sDm, long sDb,
                      const float* __restrict__ bias, int biasMode,
                      float scale, int doGelu, int M, int K)
{
    __shared__ bf16 As[2][128][40];    // [m][k], 80B pitch (16B aligned)
    __shared__ bf16 Bst[2][128][40];   // [n][k] transposed

    const int tid  = threadIdx.x;
    const int lane = tid & 31;
    const int wave = tid >> 5;
    const int wm   = wave >> 2;          // 0..1
    const int wn   = wave & 3;           // 0..3
    const int m0   = blockIdx.x * 128;
    const int n0   = blockIdx.y * 128;
    const int bat  = blockIdx.z;
    const int rsel = lane >> 4;          // 16-lane half
    const int lidx = lane & 15;

    const bf16* Ab = A  + (long)bat * sAb;
    const bf16* Bb = Bm + (long)bat * sBb;

    v8f acc[4][2];
    #pragma unroll
    for (int mi = 0; mi < 4; ++mi)
        #pragma unroll
        for (int ni = 0; ni < 2; ++ni)
            #pragma unroll
            for (int r = 0; r < 8; ++r) acc[mi][ni][r] = 0.0f;

    // global->LDS load mapping
    const int aRow = tid >> 1, aCol = (tid & 1) * 16;   // A: 2 thr/row, 16 bf16
    const int bK   = tid >> 3, bN   = (tid & 7) * 16;   // B: 16 bf16 along n
    int gr = m0 + aRow; if (gr > M - 1) gr = M - 1;     // clamp M edge (dec1)
    const long aGR = (long)gr;

    const int nK = K >> 5;

    auto stageA = [&](int kt, int buf) {
        const bf16* src = Ab + aGR * sAm + (long)kt * 32 + aCol;
        ASYNC_CP16(&As[buf][aRow][aCol],     src);
        ASYNC_CP16(&As[buf][aRow][aCol + 8], src + 8);
    };
    auto stageB = [&](int kt, int buf) {
        union { float4 f4[2]; bf16 e[16]; } bb;
        const float4* bg = reinterpret_cast<const float4*>(
            Bb + (long)(kt * 32 + bK) * sBk + n0 + bN);
        bb.f4[0] = bg[0]; bb.f4[1] = bg[1];
        #pragma unroll
        for (int i = 0; i < 16; ++i) Bst[buf][bN + i][bK] = bb.e[i];
    };

    stageA(0, 0);
    stageB(0, 0);

    for (int kt = 0; kt < nK; ++kt) {
        __syncthreads();   // all waves done with compute(kt-1) before restaging
        if (kt + 2 < nK) { // lowers to global_prefetch_b8
            __builtin_prefetch(Ab + aGR * sAm + (long)(kt + 2) * 32 + aCol, 0, 1);
            __builtin_prefetch(Bb + (long)((kt + 2) * 32 + bK) * sBk + n0 + bN, 0, 1);
        }
        if (kt + 1 < nK) {
            stageA(kt + 1, (kt + 1) & 1);
            stageB(kt + 1, (kt + 1) & 1);
            WAIT_ASYNC(2);   // 2 newer ops in flight; tile kt's 2 have landed
        } else {
            WAIT_ASYNC(0);
        }
        __syncthreads();

        const int cur = kt & 1;
        AFrag aF[4], bF[2];
        #pragma unroll
        for (int mi = 0; mi < 4; ++mi) {
            int r  = wm * 64 + mi * 16 + lidx;
            int b0 = rsel * 8;  // lanes 0-15: K {0..7,16..23}; 16-31: {8..15,24..31}
            aF[mi].h[0] = *reinterpret_cast<const v8bf*>(&As[cur][r][b0]);
            aF[mi].h[1] = *reinterpret_cast<const v8bf*>(&As[cur][r][b0 + 16]);
        }
        #pragma unroll
        for (int ni = 0; ni < 2; ++ni) {
            int c  = wn * 32 + ni * 16 + lidx;
            int kh = rsel * 8;
            bF[ni].h[0] = *reinterpret_cast<const v8bf*>(&Bst[cur][c][kh]);
            bF[ni].h[1] = *reinterpret_cast<const v8bf*>(&Bst[cur][c][kh + 16]);
        }
        #pragma unroll
        for (int mi = 0; mi < 4; ++mi)
            #pragma unroll
            for (int ni = 0; ni < 2; ++ni)
                acc[mi][ni] = __builtin_amdgcn_wmma_f32_16x16x32_bf16(
                    false, aF[mi].v, false, bF[ni].v, (short)0, acc[mi][ni], false, false);
    }

    // epilogue: scale, bias (per-col or per-row), optional exact GELU
    #pragma unroll
    for (int mi = 0; mi < 4; ++mi) {
        #pragma unroll
        for (int ni = 0; ni < 2; ++ni) {
            int col = n0 + wn * 32 + ni * 16 + lidx;
            #pragma unroll
            for (int r = 0; r < 8; ++r) {
                int row = m0 + wm * 64 + mi * 16 + rsel * 8 + r;
                if (row < M) {
                    float v = acc[mi][ni][r] * scale;
                    if (biasMode == 1) v += bias[col];
                    else if (biasMode == 2) v += bias[row];
                    if (doGelu) v = 0.5f * v * (1.0f + erff(v * 0.70710678118f));
                    long off = (long)bat * sDb + (long)row * sDm + col;
                    if (Df) Df[off] = v;
                    if (Dh) Dh[off] = f2b(v);
                }
            }
        }
    }
}

// ---------------------------------------------------------------------------
// Flash attention: grid (L/128, B*H), 256 threads.  Each wave owns 16 q-rows,
// streams 128-key K/V tiles through LDS (K async), online softmax, O in f32.
// qkv layout [B, L, 1536] (q|k|v per row); out o[B, L, 512] bf16.
// ---------------------------------------------------------------------------
__global__ __launch_bounds__(256)
void flash_attn_kernel(const bf16* __restrict__ qkv, bf16* __restrict__ o, float scale)
{
    __shared__ bf16 Ks[128][72];       // [key][k]   (B frag: contiguous k reads)
    __shared__ bf16 Vt[64][136];       // [dh][key]  (transposed V)
    __shared__ bf16 Ps[8][16][136];    // per-wave P tile [qrow][key]

    const int tid = threadIdx.x, lane = tid & 31, w = tid >> 5;
    const int bh = blockIdx.y, b = bh >> 3, h = bh & 7;
    const int qBase = blockIdx.x * 128;
    const int rsel = lane >> 4, lidx = lane & 15;
    const long ldq = 1536;

    // Q fragments (kept in registers for the whole pass)
    AFrag qf[2];
    {
        int qRow = qBase + w * 16 + lidx;
        const bf16* qp = qkv + (long)(b * NL + qRow) * ldq + h * 64;
        #pragma unroll
        for (int s = 0; s < 2; ++s) {
            int k0 = s * 32 + rsel * 8;
            qf[s].h[0] = *reinterpret_cast<const v8bf*>(qp + k0);
            qf[s].h[1] = *reinterpret_cast<const v8bf*>(qp + k0 + 16);
        }
    }

    float mrow[8], lrow[8];
    v8f oAcc[4];
    #pragma unroll
    for (int r = 0; r < 8; ++r) { mrow[r] = -3.0e38f; lrow[r] = 0.0f; }
    #pragma unroll
    for (int dt = 0; dt < 4; ++dt)
        #pragma unroll
        for (int r = 0; r < 8; ++r) oAcc[dt][r] = 0.0f;

    const int ldr = tid >> 1, ldc = (tid & 1) * 32;
    for (int kt = 0; kt < NL / 128; ++kt) {
        __syncthreads();
        {   // stage K (async, row-major) and V (sync, transposed)
            const bf16* kp = qkv + (long)(b * NL + kt * 128 + ldr) * ldq + 512 + h * 64 + ldc;
            #pragma unroll
            for (int i = 0; i < 4; ++i)
                ASYNC_CP16(&Ks[ldr][ldc + i * 8], kp + i * 8);
            union { float4 f4[4]; bf16 e[32]; } vb;
            const float4* vg = reinterpret_cast<const float4*>(kp + 512);
            vb.f4[0] = vg[0]; vb.f4[1] = vg[1]; vb.f4[2] = vg[2]; vb.f4[3] = vg[3];
            #pragma unroll
            for (int i = 0; i < 32; ++i) Vt[ldc + i][ldr] = vb.e[i];
            WAIT_ASYNC(0);
        }
        __syncthreads();

        // S = Q K^T  (8 n-tiles of 16 keys)
        v8f sAcc[8];
        #pragma unroll
        for (int nt = 0; nt < 8; ++nt) {
            AFrag kf0, kf1;
            int key = nt * 16 + lidx;
            int kh  = rsel * 8;
            kf0.h[0] = *reinterpret_cast<const v8bf*>(&Ks[key][kh]);
            kf0.h[1] = *reinterpret_cast<const v8bf*>(&Ks[key][kh + 16]);
            kf1.h[0] = *reinterpret_cast<const v8bf*>(&Ks[key][32 + kh]);
            kf1.h[1] = *reinterpret_cast<const v8bf*>(&Ks[key][32 + kh + 16]);
            v8f z;
            #pragma unroll
            for (int r = 0; r < 8; ++r) z[r] = 0.0f;
            z = __builtin_amdgcn_wmma_f32_16x16x32_bf16(false, qf[0].v, false, kf0.v, (short)0, z, false, false);
            z = __builtin_amdgcn_wmma_f32_16x16x32_bf16(false, qf[1].v, false, kf1.v, (short)0, z, false, false);
            sAcc[nt] = z;
        }

        // online softmax (row reductions across the 16-lane half via shfl_xor)
        #pragma unroll
        for (int r = 0; r < 8; ++r) {
            float mx = -3.0e38f;
            #pragma unroll
            for (int nt = 0; nt < 8; ++nt) mx = fmaxf(mx, sAcc[nt][r] * scale);
            mx = fmaxf(mx, __shfl_xor(mx, 1, 32));
            mx = fmaxf(mx, __shfl_xor(mx, 2, 32));
            mx = fmaxf(mx, __shfl_xor(mx, 4, 32));
            mx = fmaxf(mx, __shfl_xor(mx, 8, 32));
            float mnew  = fmaxf(mrow[r], mx);
            float alpha = __expf(mrow[r] - mnew);
            float rs = 0.0f;
            #pragma unroll
            for (int nt = 0; nt < 8; ++nt) {
                float p = __expf(sAcc[nt][r] * scale - mnew);
                rs += p;
                Ps[w][rsel * 8 + r][nt * 16 + lidx] = f2b(p);
            }
            rs += __shfl_xor(rs, 1, 32);
            rs += __shfl_xor(rs, 2, 32);
            rs += __shfl_xor(rs, 4, 32);
            rs += __shfl_xor(rs, 8, 32);
            lrow[r] = lrow[r] * alpha + rs;
            mrow[r] = mnew;
            #pragma unroll
            for (int dt = 0; dt < 4; ++dt) oAcc[dt][r] *= alpha;
        }

        // O += P * V   (P re-read in A-frag layout; same-wave LDS is in-order)
        #pragma unroll
        for (int f = 0; f < 4; ++f) {
            AFrag pf;
            int b0 = f * 32 + rsel * 8;
            pf.h[0] = *reinterpret_cast<const v8bf*>(&Ps[w][lidx][b0]);
            pf.h[1] = *reinterpret_cast<const v8bf*>(&Ps[w][lidx][b0 + 16]);
            #pragma unroll
            for (int dt = 0; dt < 4; ++dt) {
                AFrag vf;
                int col = dt * 16 + lidx;
                int kh  = f * 32 + rsel * 8;
                vf.h[0] = *reinterpret_cast<const v8bf*>(&Vt[col][kh]);
                vf.h[1] = *reinterpret_cast<const v8bf*>(&Vt[col][kh + 16]);
                oAcc[dt] = __builtin_amdgcn_wmma_f32_16x16x32_bf16(
                    false, pf.v, false, vf.v, (short)0, oAcc[dt], false, false);
            }
        }
    }

    #pragma unroll
    for (int dt = 0; dt < 4; ++dt)
        #pragma unroll
        for (int r = 0; r < 8; ++r) {
            int q = qBase + w * 16 + rsel * 8 + r;
            float v = oAcc[dt][r] / lrow[r];
            o[(long)(b * NL + q) * ND + h * 64 + dt * 16 + lidx] = f2b(v);
        }
}

// ---------------------------------------------------------------------------
// LayerNorm over D=512: y = LN(x [+ res]) * g + b, emits f32 and/or bf16.
// ---------------------------------------------------------------------------
__global__ __launch_bounds__(256)
void layernorm_kernel(const float* __restrict__ x, const float* __restrict__ res,
                      const float* __restrict__ g, const float* __restrict__ bta,
                      float* __restrict__ yf, bf16* __restrict__ yh)
{
    __shared__ float r1[256], r2[256];
    long row = blockIdx.x;
    int tid = threadIdx.x;
    const float* xr = x + row * ND;
    float v0 = xr[tid]       + (res ? res[row * ND + tid]       : 0.0f);
    float v1 = xr[tid + 256] + (res ? res[row * ND + tid + 256] : 0.0f);
    r1[tid] = v0 + v1;
    r2[tid] = v0 * v0 + v1 * v1;
    __syncthreads();
    for (int off = 128; off > 0; off >>= 1) {
        if (tid < off) { r1[tid] += r1[tid + off]; r2[tid] += r2[tid + off]; }
        __syncthreads();
    }
    float mean = r1[0] * (1.0f / ND);
    float var  = r2[0] * (1.0f / ND) - mean * mean;
    float rstd = rsqrtf(var + 1e-5f);
    float o0 = (v0 - mean) * rstd * g[tid]       + bta[tid];
    float o1 = (v1 - mean) * rstd * g[tid + 256] + bta[tid + 256];
    if (yf) { yf[row * ND + tid] = o0; yf[row * ND + tid + 256] = o1; }
    if (yh) { yh[row * ND + tid] = f2b(o0); yh[row * ND + tid + 256] = f2b(o1); }
}

// ---------------------------------------------------------------------------
// RevIN stats: per (b,c) mean/std over L (biased var).
// ---------------------------------------------------------------------------
__global__ __launch_bounds__(256)
void revin_stats_kernel(const float* __restrict__ x, float* __restrict__ meanA,
                        float* __restrict__ stdA)
{
    int bc = blockIdx.x, b = bc / NC, c = bc % NC;
    __shared__ float r1[256], r2[256];
    int tid = threadIdx.x;
    float s = 0.0f, s2 = 0.0f;
    for (int l = tid; l < NL; l += 256) {
        float v = x[((long)b * NL + l) * NC + c];
        s += v; s2 += v * v;
    }
    r1[tid] = s; r2[tid] = s2; __syncthreads();
    for (int off = 128; off > 0; off >>= 1) {
        if (tid < off) { r1[tid] += r1[tid + off]; r2[tid] += r2[tid + off]; }
        __syncthreads();
    }
    if (tid == 0) {
        float m = r1[0] * (1.0f / NL);
        float var = r2[0] * (1.0f / NL) - m * m;
        meanA[bc] = m;
        stdA[bc] = sqrtf(var + 1e-5f);
    }
}

// ---------------------------------------------------------------------------
// RevIN-normalize + circular Conv1d(C->D, k=3) + sinusoidal PE.  Block (l,b).
// ---------------------------------------------------------------------------
__global__ __launch_bounds__(256)
void embed_kernel(const float* __restrict__ x, const float* __restrict__ meanA,
                  const float* __restrict__ stdA, const float* __restrict__ rw,
                  const float* __restrict__ rb, const float* __restrict__ tok,
                  float* __restrict__ hf, bf16* __restrict__ hb)
{
    int l = blockIdx.x, b = blockIdx.y;
    __shared__ float xn[64];
    int tid = threadIdx.x;
    if (tid < 63) {
        int k = tid / NC, c = tid % NC;
        int li = l + k - 1;
        if (li < 0) li += NL;
        if (li >= NL) li -= NL;
        float v = x[((long)b * NL + li) * NC + c];
        xn[tid] = (v - meanA[b * NC + c]) / stdA[b * NC + c] * rw[c] + rb[c];
    }
    __syncthreads();
    #pragma unroll
    for (int dd = 0; dd < 2; ++dd) {
        int d = tid + dd * 256;
        float acc = 0.0f;
        for (int j = 0; j < 63; ++j) acc += xn[j] * tok[(long)j * ND + d];
        int de = d & ~1;
        float div = __expf((float)de * (-9.210340372f / (float)ND));  // ln(10000)
        acc += (d & 1) ? __cosf((float)l * div) : __sinf((float)l * div);
        long off = ((long)b * NL + l) * ND + d;
        hf[off] = acc;
        hb[off] = f2b(acc);
    }
}

// ---------------------------------------------------------------------------
// Decoder tail: out[b,p,c] = denorm( mid[b,p,:] @ linW[:,c] + linB[c] )
// ---------------------------------------------------------------------------
__global__ __launch_bounds__(256)
void dec_out_kernel(const float* __restrict__ mid, const float* __restrict__ Wl,
                    const float* __restrict__ bl, const float* __restrict__ rw,
                    const float* __restrict__ rb, const float* __restrict__ meanA,
                    const float* __restrict__ stdA, float* __restrict__ out)
{
    int p = blockIdx.x, b = blockIdx.y;
    __shared__ float rowv[ND];
    int tid = threadIdx.x;
    const float* mr = mid + ((long)b * NP + p) * ND;
    rowv[tid] = mr[tid];
    rowv[tid + 256] = mr[tid + 256];
    __syncthreads();
    int c = tid >> 3, j = tid & 7;
    if (c < NC) {
        float acc = 0.0f;
        for (int k = j; k < ND; k += 8) acc += rowv[k] * Wl[(long)k * NC + c];
        acc += __shfl_xor(acc, 1, 32);
        acc += __shfl_xor(acc, 2, 32);
        acc += __shfl_xor(acc, 4, 32);
        if (j == 0) {
            float v = acc + bl[c];
            v = (v - rb[c]) / (rw[c] + 1e-10f) * stdA[b * NC + c] + meanA[b * NC + c];
            out[((long)b * NP + p) * NC + c] = v;
        }
    }
}

// f32 -> bf16 strided 2D conversion (weight repack)
__global__ void cvt_bf16_kernel(const float* __restrict__ src, long srcLd,
                                bf16* __restrict__ dst, long dstLd, int rows, int cols)
{
    long idx = (long)blockIdx.x * 256 + threadIdx.x;
    long total = (long)rows * cols;
    if (idx >= total) return;
    long r = idx / cols, c = idx % cols;
    dst[r * dstLd + c] = f2b(src[r * srcLd + c]);
}

// ---------------------------------------------------------------------------
// Orchestration
// ---------------------------------------------------------------------------
extern "C" void kernel_launch(void* const* d_in, const int* in_sizes, int n_in,
                              void* d_out, int out_size, void* d_ws, size_t ws_size,
                              hipStream_t stream)
{
    (void)in_sizes; (void)n_in; (void)out_size; (void)ws_size;
    const float* x_enc = (const float*)d_in[0];
    const float* rvw  = (const float*)d_in[1];
    const float* rvb  = (const float*)d_in[2];
    const float* tok  = (const float*)d_in[3];
    const float* Wq = (const float*)d_in[4];  const float* bq = (const float*)d_in[5];
    const float* Wk = (const float*)d_in[6];  const float* bk = (const float*)d_in[7];
    const float* Wv = (const float*)d_in[8];  const float* bv = (const float*)d_in[9];
    const float* Wo = (const float*)d_in[10]; const float* bo = (const float*)d_in[11];
    const float* W1 = (const float*)d_in[12]; const float* b1 = (const float*)d_in[13];
    const float* W2 = (const float*)d_in[14]; const float* b2 = (const float*)d_in[15];
    const float* g1 = (const float*)d_in[16]; const float* be1 = (const float*)d_in[17];
    const float* g2 = (const float*)d_in[18]; const float* be2 = (const float*)d_in[19];
    const float* ng = (const float*)d_in[20]; const float* nb = (const float*)d_in[21];
    const float* decW = (const float*)d_in[22]; const float* decB = (const float*)d_in[23];
    const float* linW = (const float*)d_in[24]; const float* linB = (const float*)d_in[25];
    float* out = (float*)d_out;

    char* base = (char*)d_ws;
    size_t off = 0;
    auto alloc = [&](size_t bytes) -> void* {
        off = (off + 255) & ~(size_t)255;
        void* p = (void*)(base + off);
        off += bytes;
        return p;
    };

    const long NT = (long)NB * NL;  // 32768 tokens
    bf16* Wqkv_bf = (bf16*)alloc((size_t)NE * 512 * 1536 * 2);
    bf16* Wo_bf   = (bf16*)alloc((size_t)NE * 512 * 512 * 2);
    bf16* W1_bf   = (bf16*)alloc((size_t)NE * 512 * 2048 * 2);
    bf16* W2_bf   = (bf16*)alloc((size_t)NE * 2048 * 512 * 2);
    bf16* decW_bf = (bf16*)alloc((size_t)NP * NL * 2);
    bf16* h_bf    = (bf16*)alloc((size_t)NT * ND * 2);
    bf16* qkv_bf  = (bf16*)alloc((size_t)NT * 1536 * 2);
    bf16* o_bf    = (bf16*)alloc((size_t)NT * ND * 2);
    bf16* x1_bf   = (bf16*)alloc((size_t)NT * ND * 2);
    bf16* ffn_bf  = (bf16*)alloc((size_t)NT * NDF * 2);
    float* bqkv   = (float*)alloc((size_t)NE * 1536 * 4);
    float* meanA  = (float*)alloc((size_t)NB * NC * 4);
    float* stdA   = (float*)alloc((size_t)NB * NC * 4);
    float* h_f    = (float*)alloc((size_t)NT * ND * 4);
    float* t_f    = (float*)alloc((size_t)NT * ND * 4);   // attn-out / ffn2-out
    float* x1_f   = (float*)alloc((size_t)NT * ND * 4);
    float* mid_f  = (float*)alloc((size_t)NB * NP * ND * 4);

    auto cvt = [&](const float* s, long sld, bf16* d, long dld, int rows, int cols) {
        long tot = (long)rows * cols;
        int grid = (int)((tot + 255) / 256);
        cvt_bf16_kernel<<<grid, 256, 0, stream>>>(s, sld, d, dld, rows, cols);
    };

    // One-time weight repack to bf16 (QKV fused into [512,1536])
    for (int l = 0; l < NE; ++l) {
        cvt(Wq + (long)l * 512 * 512, 512, Wqkv_bf + (long)l * 512 * 1536 + 0,    1536, 512, 512);
        cvt(Wk + (long)l * 512 * 512, 512, Wqkv_bf + (long)l * 512 * 1536 + 512,  1536, 512, 512);
        cvt(Wv + (long)l * 512 * 512, 512, Wqkv_bf + (long)l * 512 * 1536 + 1024, 1536, 512, 512);
        cvt(Wo + (long)l * 512 * 512, 512, Wo_bf + (long)l * 512 * 512, 512, 512, 512);
        cvt(W1 + (long)l * 512 * 2048, 2048, W1_bf + (long)l * 512 * 2048, 2048, 512, 2048);
        cvt(W2 + (long)l * 2048 * 512, 512, W2_bf + (long)l * 2048 * 512, 512, 2048, 512);
        (void)hipMemcpyAsync(bqkv + l * 1536,        bq + l * 512, 512 * 4, hipMemcpyDeviceToDevice, stream);
        (void)hipMemcpyAsync(bqkv + l * 1536 + 512,  bk + l * 512, 512 * 4, hipMemcpyDeviceToDevice, stream);
        (void)hipMemcpyAsync(bqkv + l * 1536 + 1024, bv + l * 512, 512 * 4, hipMemcpyDeviceToDevice, stream);
    }
    cvt(decW, NL, decW_bf, NL, NP, NL);

    revin_stats_kernel<<<NB * NC, 256, 0, stream>>>(x_enc, meanA, stdA);
    embed_kernel<<<dim3(NL, NB), 256, 0, stream>>>(x_enc, meanA, stdA, rvw, rvb, tok, h_f, h_bf);

    auto gemm = [&](const bf16* A, long sAm, long sAb,
                    const bf16* Bm, long sBk, long sBb,
                    float* Df, bf16* Dh, long sDm, long sDb,
                    const float* bias, int biasMode, float scale, int gelu,
                    int M, int N, int K, int batch) {
        dim3 grid((M + 127) / 128, N / 128, batch);
        gemm_bf16_kernel<<<grid, 256, 0, stream>>>(A, sAm, sAb, Bm, sBk, sBb,
                                                   Df, Dh, sDm, sDb, bias, biasMode,
                                                   scale, gelu, M, K);
    };

    for (int l = 0; l < NE; ++l) {
        // fused QKV projection -> bf16 [NT, 1536]
        gemm(h_bf, ND, 0, Wqkv_bf + (long)l * 512 * 1536, 1536, 0,
             nullptr, qkv_bf, 1536, 0, bqkv + l * 1536, 1, 1.0f, 0, (int)NT, 1536, ND, 1);
        // flash attention -> o_bf [NT, 512]
        flash_attn_kernel<<<dim3(NL / 128, NB * NH), 256, 0, stream>>>(qkv_bf, o_bf, 0.125f);
        // output projection -> f32
        gemm(o_bf, ND, 0, Wo_bf + (long)l * 512 * 512, 512, 0,
             t_f, nullptr, ND, 0, bo + l * ND, 1, 1.0f, 0, (int)NT, ND, ND, 1);
        // x1 = LN(h + attn_out)
        layernorm_kernel<<<(int)NT, 256, 0, stream>>>(t_f, h_f, g1 + l * ND, be1 + l * ND, x1_f, x1_bf);
        // FFN1 + GELU -> bf16 [NT, 2048]
        gemm(x1_bf, ND, 0, W1_bf + (long)l * 512 * 2048, 2048, 0,
             nullptr, ffn_bf, NDF, 0, b1 + l * NDF, 1, 1.0f, 1, (int)NT, NDF, ND, 1);
        // FFN2 -> f32
        gemm(ffn_bf, NDF, 0, W2_bf + (long)l * 2048 * 512, 512, 0,
             t_f, nullptr, ND, 0, b2 + l * ND, 1, 1.0f, 0, (int)NT, ND, NDF, 1);
        // h = LN(x1 + ffn_out)
        layernorm_kernel<<<(int)NT, 256, 0, stream>>>(t_f, x1_f, g2 + l * ND, be2 + l * ND, h_f, h_bf);
    }

    // final LN (bf16 only, feeds decoder GEMM)
    layernorm_kernel<<<(int)NT, 256, 0, stream>>>(h_f, nullptr, ng, nb, nullptr, h_bf);

    // decoder seq-mix: mid[b,p,d] = decW[p,:] @ h[b,:,d] + decB[p]  (batched)
    gemm(decW_bf, NL, 0, h_bf, ND, (long)NL * ND,
         mid_f, nullptr, ND, (long)NP * ND, decB, 2, 1.0f, 0, NP, ND, NL, NB);

    // D->C linear + RevIN denorm
    dec_out_kernel<<<dim3(NP, NB), 256, 0, stream>>>(mid_f, linW, linB, rvw, rvb, meanA, stdA, out);
}
